// RGATLayer_26723286515872
// MI455X (gfx1250) — matrix operand order repaired and checked by reading, not compiled, not run on hardware
//
#include <hip/hip_runtime.h>

typedef __attribute__((ext_vector_type(16))) _Float16 v16h;
typedef __attribute__((ext_vector_type(8)))  _Float16 v8h;
typedef __attribute__((ext_vector_type(4)))  _Float16 v4h;
typedef __attribute__((ext_vector_type(8)))  float    v8f;

#define DFEAT 128
#define BSTRIDE 136   // padded LDS row stride (halves): 272B rows, 16B aligned, bank-friendly

// ---------------------------------------------------------------------------
// Kernel 0: zero the output accumulator and the per-relation softmax state.
// ---------------------------------------------------------------------------
__global__ void rgat_init(float* __restrict__ out, size_t n,
                          unsigned* __restrict__ segmax_u, float* __restrict__ segsum) {
  size_t i = (size_t)blockIdx.x * blockDim.x + threadIdx.x;
  size_t stride = (size_t)gridDim.x * blockDim.x;
  for (; i < n; i += stride) out[i] = 0.0f;
  if (blockIdx.x == 0 && threadIdx.x < 8) {
    segmax_u[threadIdx.x] = 0u;   // encoded identity for order-preserving fmax
    segsum[threadIdx.x]   = 0.0f;
  }
}

// ---------------------------------------------------------------------------
// Kernel 1: h[r] = x @ W[r], f32 accumulate via WMMA f16, stored as f16.
// grid = (ceil(N/128), R), block = 256 (8 waves). Wave w owns rows
// [blockIdx.x*128 + w*16, +16) x all 128 output columns (8 n-tiles).
// ---------------------------------------------------------------------------
__global__ void rgat_gemm(const float* __restrict__ x, const float* __restrict__ W,
                          _Float16* __restrict__ h, int Nnodes) {
  __shared__ _Float16 Bs[DFEAT * BSTRIDE];   // transposed: Bs[n][k] = W[r][k][n]
  const int r = blockIdx.y;
  const float* Wr = W + (size_t)r * DFEAT * DFEAT;

  // Stage W[r] (f32, row-major [k][n]) into LDS transposed as f16.
  for (int idx = threadIdx.x; idx < DFEAT * DFEAT; idx += blockDim.x) {
    int k = idx >> 7, n = idx & 127;
    Bs[n * BSTRIDE + k] = (_Float16)Wr[idx];
  }
  __syncthreads();

  const int wave = threadIdx.x >> 5;
  const int lane = threadIdx.x & 31;
  const int m    = lane & 15;
  const int hiHalf = lane >> 4;            // 0 or 1
  const int kbA  = hiHalf << 3;            // A-frag K base: 0 or 8 (interleaved layout)
  const int kbB  = hiHalf << 3;            // B-frag K base: 0 or 8 (interleaved layout)
  const int row0 = blockIdx.x * 128 + wave * 16;
  const bool fullTile = (row0 + 16 <= Nnodes);   // wave-uniform
  const int row  = row0 + m;
  const float* xrow = x + (size_t)row * DFEAT;

  // Load & convert the A fragments for all four K-chunks (K = 128 = 4 x 32).
  v16h a[4];
  #pragma unroll
  for (int kc = 0; kc < 4; ++kc) {
    int base = kc * 32 + kbA;
    float4 f0 = {0,0,0,0}, f1 = {0,0,0,0}, f2 = {0,0,0,0}, f3 = {0,0,0,0};
    if (fullTile || row < Nnodes) {
      f0 = *(const float4*)(xrow + base);        // K = kbA + 0..3
      f1 = *(const float4*)(xrow + base + 4);    // K = kbA + 4..7
      f2 = *(const float4*)(xrow + base + 16);   // K = kbA + 16..19
      f3 = *(const float4*)(xrow + base + 20);   // K = kbA + 20..23
    }
    a[kc][0]  = (_Float16)f0.x; a[kc][1]  = (_Float16)f0.y;
    a[kc][2]  = (_Float16)f0.z; a[kc][3]  = (_Float16)f0.w;
    a[kc][4]  = (_Float16)f1.x; a[kc][5]  = (_Float16)f1.y;
    a[kc][6]  = (_Float16)f1.z; a[kc][7]  = (_Float16)f1.w;
    a[kc][8]  = (_Float16)f2.x; a[kc][9]  = (_Float16)f2.y;
    a[kc][10] = (_Float16)f2.z; a[kc][11] = (_Float16)f2.w;
    a[kc][12] = (_Float16)f3.x; a[kc][13] = (_Float16)f3.y;
    a[kc][14] = (_Float16)f3.z; a[kc][15] = (_Float16)f3.w;
  }

  // D layout: VGPR v -> row (row0 + hiHalf*8 + v), col = nt*16 + (lane & 15).
  // Single invariant base pointer => stores use immediate offsets.
  const size_t hbase = (size_t)r * Nnodes * DFEAT;
  const int rbase = row0 + (hiHalf << 3);
  _Float16* __restrict__ hp = h + hbase + (size_t)rbase * DFEAT + m;

  #pragma unroll
  for (int nt = 0; nt < 8; ++nt) {
    const int n = nt * 16 + m;
    v8f acc = {};
    #pragma unroll
    for (int kc = 0; kc < 4; ++kc) {
      const _Float16* bp = &Bs[n * BSTRIDE + kc * 32 + kbB];
      v8h b0 = *(const v8h*)bp;        // K = kbB + 0..7
      v8h b1 = *(const v8h*)(bp + 16); // K = kbB + 16..23
      v16h b;
      #pragma unroll
      for (int i = 0; i < 8; ++i) { b[i] = b0[i]; b[8 + i] = b1[i]; }
      acc = __builtin_amdgcn_wmma_f32_16x16x32_f16(
          /*neg_a=*/false, a[kc], /*neg_b=*/false, b,
          /*c_mod=*/(short)0, acc, /*reuse_a=*/false, /*reuse_b=*/false);
    }
    if (fullTile) {
      #pragma unroll
      for (int v = 0; v < 8; ++v)
        hp[(size_t)v * DFEAT + nt * 16] = (_Float16)acc[v];
    } else {
      #pragma unroll
      for (int v = 0; v < 8; ++v)
        if (rbase + v < Nnodes)
          hp[(size_t)v * DFEAT + nt * 16] = (_Float16)acc[v];
    }
  }
}

// ---------------------------------------------------------------------------
// Kernel 2: per-edge attention logit + per-relation running max.
// One wave per edge; block-local LDS max, then 8 global atomics per block.
// ---------------------------------------------------------------------------
__global__ void rgat_edge_dot(const _Float16* __restrict__ h, const int* __restrict__ ei,
                              const int* __restrict__ et, float* __restrict__ attn,
                              unsigned* __restrict__ segmax_u, int E, int Nnodes) {
  __shared__ unsigned smax[8];
  if (threadIdx.x < 8) smax[threadIdx.x] = 0u;
  __syncthreads();

  const int lane = threadIdx.x & 31;
  const int e = blockIdx.x * (blockDim.x >> 5) + (threadIdx.x >> 5);
  if (e < E) {
    const int rel = et[e];
    const int src = ei[e];
    const int dst = ei[E + e];
    const v4h vs = *(const v4h*)(h + (((size_t)rel * Nnodes + src) << 7) + lane * 4);
    const v4h vd = *(const v4h*)(h + (((size_t)rel * Nnodes + dst) << 7) + lane * 4);
    float p = (float)vs[0] * (float)vd[0] + (float)vs[1] * (float)vd[1]
            + (float)vs[2] * (float)vd[2] + (float)vs[3] * (float)vd[3];
    #pragma unroll
    for (int off = 16; off > 0; off >>= 1) p += __shfl_xor(p, off, 32);
    if (lane == 0) {
      float aa = p > 0.0f ? p : 0.2f * p;     // leaky_relu(0.2)
      attn[e] = aa;
      unsigned ub  = __float_as_uint(aa);
      unsigned enc = (ub & 0x80000000u) ? ~ub : (ub | 0x80000000u); // order-preserving
      atomicMax(&smax[rel], enc);
    }
  }
  __syncthreads();
  if (threadIdx.x < 8 && smax[threadIdx.x] != 0u)
    atomicMax(&segmax_u[threadIdx.x], smax[threadIdx.x]);
}

// ---------------------------------------------------------------------------
// Kernel 3: ex = exp(attn - segmax[rel]) in place; per-relation sum via LDS.
// ---------------------------------------------------------------------------
__global__ void rgat_edge_exp(const int* __restrict__ et, const unsigned* __restrict__ segmax_u,
                              float* __restrict__ attn, float* __restrict__ segsum, int E) {
  __shared__ float ssum[8];
  if (threadIdx.x < 8) ssum[threadIdx.x] = 0.0f;
  __syncthreads();
  const int e = blockIdx.x * blockDim.x + threadIdx.x;
  if (e < E) {
    const int rel = et[e];
    unsigned u = segmax_u[rel];
    float mx = (u & 0x80000000u) ? __uint_as_float(u & 0x7FFFFFFFu) : __uint_as_float(~u);
    float ex = __expf(attn[e] - mx);
    attn[e] = ex;
    atomicAdd(&ssum[rel], ex);
  }
  __syncthreads();
  if (threadIdx.x < 8 && ssum[threadIdx.x] != 0.0f)
    atomicAdd(&segsum[threadIdx.x], ssum[threadIdx.x]);
}

// ---------------------------------------------------------------------------
// Kernel 4: out[dst] += h_src * (ex / segsum[rel]).  One wave per edge,
// 4 global f32 atomic adds per lane.
// ---------------------------------------------------------------------------
__global__ void rgat_edge_scatter(const _Float16* __restrict__ h, const int* __restrict__ ei,
                                  const int* __restrict__ et, const float* __restrict__ attn,
                                  const float* __restrict__ segsum, float* __restrict__ out,
                                  int E, int Nnodes) {
  const int e = blockIdx.x * (blockDim.x >> 5) + (threadIdx.x >> 5);
  if (e >= E) return;
  const int lane = threadIdx.x & 31;
  const int rel = et[e];
  const int src = ei[e];
  const int dst = ei[E + e];
  const float a = attn[e] / segsum[rel];
  const v4h vs = *(const v4h*)(h + (((size_t)rel * Nnodes + src) << 7) + lane * 4);
  float* op = out + ((size_t)dst << 7) + lane * 4;
  atomicAdd(op + 0, (float)vs[0] * a);
  atomicAdd(op + 1, (float)vs[1] * a);
  atomicAdd(op + 2, (float)vs[2] * a);
  atomicAdd(op + 3, (float)vs[3] * a);
}

// ---------------------------------------------------------------------------
extern "C" void kernel_launch(void* const* d_in, const int* in_sizes, int n_in,
                              void* d_out, int out_size, void* d_ws, size_t ws_size,
                              hipStream_t stream) {
  const float* x = (const float*)d_in[0];
  const float* W = (const float*)d_in[1];
  const int*   ei = (const int*)d_in[2];
  const int*   et = (const int*)d_in[3];
  float* out = (float*)d_out;

  const int Nnodes = in_sizes[0] / DFEAT;             // 50000
  const int R      = in_sizes[1] / (DFEAT * DFEAT);   // 8
  const int E      = in_sizes[3];                     // 600000

  // Workspace layout: h (f16, R*N*128) | attn (f32, E) | segmax (u32 x8) | segsum (f32 x8)
  _Float16* h = (_Float16*)d_ws;
  size_t hbytes = ((size_t)R * Nnodes * DFEAT * sizeof(_Float16) + 255) & ~(size_t)255;
  float*    attn   = (float*)((char*)d_ws + hbytes);
  unsigned* segmax = (unsigned*)(attn + E);
  float*    segsum = (float*)(segmax + 8);

  rgat_init<<<1024, 256, 0, stream>>>(out, (size_t)Nnodes * DFEAT, segmax, segsum);

  dim3 ggrid((Nnodes + 127) / 128, R);
  rgat_gemm<<<ggrid, 256, 0, stream>>>(x, W, h, Nnodes);

  int wpb = 256 / 32;  // 8 waves (edges) per block
  rgat_edge_dot<<<(E + wpb - 1) / wpb, 256, 0, stream>>>(h, ei, et, attn, segmax, E, Nnodes);
  rgat_edge_exp<<<(E + 255) / 256, 256, 0, stream>>>(et, segmax, attn, segsum, E);
  rgat_edge_scatter<<<(E + wpb - 1) / wpb, 256, 0, stream>>>(h, ei, et, attn, segsum, out, E, Nnodes);
}
